// IntMLP_70136815943786
// MI455X (gfx1250) — 8-bit matrix operand order repaired and checked by reading, compiled, not run on hardware
//
#include <hip/hip_runtime.h>
#include <cstdint>
#include <cstddef>

typedef __attribute__((ext_vector_type(8))) int v8i;

#define TOKENS 8192
#define HIDDEN 2048

#define BM 128
#define BN 256
#define BK 64
#define LDS_STRIDE 80
#define XBUF (BM * LDS_STRIDE)
#define WBUF (BN * LDS_STRIDE)

union AFrag { v8i v; int2 p[4]; };
union BFrag { v8i v; int4 q[2]; };

__global__ void IntMLP_pack_kernel(const int* __restrict__ in,
                                   signed char* __restrict__ out, int n4) {
    int i = blockIdx.x * blockDim.x + threadIdx.x;
    if (i < n4) {
        int4 v = *(const int4*)(in + (size_t)i * 4);
        char4 c;
        c.x = (signed char)v.x;
        c.y = (signed char)v.y;
        c.z = (signed char)v.z;
        c.w = (signed char)v.w;
        *(char4*)(out + (size_t)i * 4) = c;
    }
}

template <int SCALE, bool WRITE_FLOAT>
__global__ __launch_bounds__(256) void IntMLP_gemm_iu8_kernel(
    const signed char* __restrict__ A,
    const signed char* __restrict__ B,
    signed char* __restrict__ out8,
    float* __restrict__ outf,
    int M, int N, int K)
{
    __shared__ alignas(16) signed char xs[3][XBUF];
    __shared__ alignas(16) signed char wsh[3][WBUF];

    const int tid  = threadIdx.x;
    const int lane = tid & 31;
    const int wave = tid >> 5;
    const int wm   = wave >> 2;
    const int wn   = wave & 3;
    const int half = lane >> 4;
    const int lr   = lane & 15;

    const int m0 = blockIdx.y * BM;
    const int n0 = blockIdx.x * BN;

    unsigned xrel[2];
    const signed char* agp[2];
    #pragma unroll
    for (int it = 0; it < 2; ++it) {
        int c = tid + it * 256;
        int row = c >> 2, col = (c & 3) * 16;
        xrel[it] = (unsigned)(row * LDS_STRIDE + col);
        agp[it]  = A + (size_t)(m0 + row) * K + col;
    }
    unsigned wrel[4];
    const signed char* bgp[4];
    #pragma unroll
    for (int it = 0; it < 4; ++it) {
        int c = tid + it * 256;
        int row = c >> 2, col = (c & 3) * 16;
        wrel[it] = (unsigned)(row * LDS_STRIDE + col);
        bgp[it]  = B + (size_t)(n0 + row) * K + col;
    }
    const unsigned xbase = (unsigned)(uintptr_t)&xs[0][0];
    const unsigned wbase = (unsigned)(uintptr_t)&wsh[0][0];

    auto issue_tile = [&](int buf, int kkn) {
        #pragma unroll
        for (int it = 0; it < 2; ++it) {
            unsigned lds = xbase + (unsigned)buf * XBUF + xrel[it];
            const signed char* g = agp[it] + kkn;
            asm volatile("global_load_async_to_lds_b128 %0, %1, off"
                         :: "v"(lds), "v"(g) : "memory");
        }
        #pragma unroll
        for (int it = 0; it < 4; ++it) {
            unsigned lds = wbase + (unsigned)buf * WBUF + wrel[it];
            const signed char* g = bgp[it] + kkn;
            asm volatile("global_load_async_to_lds_b128 %0, %1, off"
                         :: "v"(lds), "v"(g) : "memory");
        }
    };

    v8i acc[4][4] = {};

    auto compute_step = [&](int buf) {
        const signed char* xb = &xs[buf][0];
        const signed char* wb = &wsh[buf][0];
        AFrag a[4];
        #pragma unroll
        for (int i = 0; i < 4; ++i) {
            int base = (wm * 64 + i * 16 + lr) * LDS_STRIDE + half * 16;
            a[i].p[0] = *(const int2*)&xb[base];
            a[i].p[1] = *(const int2*)&xb[base + 8];
            a[i].p[2] = *(const int2*)&xb[base + 32];
            a[i].p[3] = *(const int2*)&xb[base + 40];
        }
        BFrag b[4];
        #pragma unroll
        for (int j = 0; j < 4; ++j) {
            int base = (wn * 64 + j * 16 + lr) * LDS_STRIDE + half * 16;
            b[j].q[0] = *(const int4*)&wb[base];
            b[j].q[1] = *(const int4*)&wb[base + 32];
        }
        #pragma unroll
        for (int i = 0; i < 4; ++i)
            #pragma unroll
            for (int j = 0; j < 4; ++j)
                acc[i][j] = __builtin_amdgcn_wmma_i32_16x16x64_iu8(
                     true, a[i].v,  true, b[j].v,
                    acc[i][j],  false,  false);
    };

    const int NK = K / BK;

    issue_tile(0, 0);
    issue_tile(1, BK);

    int cs = 0;
    for (int s = 0; s < NK - 2; ++s) {
        asm volatile("s_wait_asynccnt 6" ::: "memory");
        __syncthreads();
        int ib = cs + 2; if (ib >= 3) ib -= 3;
        issue_tile(ib, (s + 2) * BK);
        compute_step(cs);
        ++cs; if (cs == 3) cs = 0;
    }
    asm volatile("s_wait_asynccnt 6" ::: "memory");
    __syncthreads();
    compute_step(cs);
    ++cs; if (cs == 3) cs = 0;
    asm volatile("s_wait_asynccnt 0" ::: "memory");
    __syncthreads();
    compute_step(cs);

    #pragma unroll
    for (int i = 0; i < 4; ++i) {
        #pragma unroll
        for (int j = 0; j < 4; ++j) {
            int colg = n0 + wn * 64 + j * 16 + lr;
            #pragma unroll
            for (int g = 0; g < 8; ++g) {
                int rowg = m0 + wm * 64 + i * 16 + half * 8 + g;
                int v = acc[i][j][g];
                unsigned t = (unsigned)(v + (SCALE << 18));
                int q = (int)(t / (unsigned)SCALE) - (1 << 18);
                q = q < -127 ? -127 : (q > 127 ? 127 : q);
                if (WRITE_FLOAT) outf[(size_t)rowg * N + colg] = (float)q;
                else             out8[(size_t)rowg * N + colg] = (signed char)q;
            }
        }
    }
}

extern "C" void kernel_launch(void* const* d_in, const int* in_sizes, int n_in,
                              void* d_out, int out_size, void* d_ws, size_t ws_size,
                              hipStream_t stream) {
    const int* x32  = (const int*)d_in[0];
    const int* w132 = (const int*)d_in[1];
    const int* w232 = (const int*)d_in[2];
    float* outf = (float*)d_out;

    const size_t xN  = (size_t)TOKENS * HIDDEN;
    const size_t w1N = (size_t)(4 * HIDDEN) * HIDDEN;
    const size_t w2N = (size_t)HIDDEN * (4 * HIDDEN);

    signed char* x8  = (signed char*)d_ws;
    signed char* w18 = x8 + xN;
    signed char* w28 = w18 + w1N;
    signed char* h8  = w28 + w2N;

    IntMLP_pack_kernel<<<(unsigned)((xN / 4 + 255) / 256), 256, 0, stream>>>(x32, x8, (int)(xN / 4));
    IntMLP_pack_kernel<<<(unsigned)((w1N / 4 + 255) / 256), 256, 0, stream>>>(w132, w18, (int)(w1N / 4));
    IntMLP_pack_kernel<<<(unsigned)((w2N / 4 + 255) / 256), 256, 0, stream>>>(w232, w28, (int)(w2N / 4));

    dim3 blk(256);
    dim3 g1((4 * HIDDEN) / BN, TOKENS / BM);
    IntMLP_gemm_iu8_kernel<720, false><<<g1, blk, 0, stream>>>(
        x8, w18, h8, nullptr, TOKENS, 4 * HIDDEN, HIDDEN);
    dim3 g2(HIDDEN / BN, TOKENS / BM);
    IntMLP_gemm_iu8_kernel<1440, true><<<g2, blk, 0, stream>>>(
        h8, w28, nullptr, outf, TOKENS, HIDDEN, 4 * HIDDEN);
}
